// OuterProductMean_53197464928697
// MI455X (gfx1250) — matrix operand order repaired and checked by reading, not compiled, not run on hardware
//
#include <hip/hip_runtime.h>

#define C_DIM 256     // DIM_MSA
#define D_SEQ 384     // DIM_SEQ
#define H_DIM 32      // DIM_HID
#define B_N   2
#define S_N   128
#define L_N   1024
#define K_OP  1024    // DIM_HID * DIM_HID

typedef __attribute__((ext_vector_type(16))) __bf16 v16bf;
typedef __attribute__((ext_vector_type(8)))  float  v8f;

union Op16 { unsigned int u[8]; v16bf v; };

// packed f32x2 -> bf16x2 (low = first arg).
// Prefer the packed-convert builtin when the toolchain declares it; otherwise
// use V_FMA_MIXLO_BF16 / V_FMA_MIXHI_BF16 (CDNA5 ISA §15.10, opcodes 62/63):
// D.half = cvt_bf16(fma(x, 1.0, 0)) -- 2 VALU per pair vs ~7 for bit tricks.
__device__ __forceinline__ unsigned int pack2bf(float lo, float hi) {
#if __has_builtin(__builtin_amdgcn_cvt_pk_bf16_f32)
  auto r = __builtin_amdgcn_cvt_pk_bf16_f32(lo, hi);
  unsigned int u;
  __builtin_memcpy(&u, &r, sizeof(u));
  return u;
#else
  unsigned int r = 0;
  asm("v_fma_mixlo_bf16 %0, %1, 1.0, 0\n\t"
      "v_fma_mixhi_bf16 %0, %2, 1.0, 0"
      : "+v"(r) : "v"(lo), "v"(hi));
  return r;
#endif
}

// A-operand (16x32 bf16, MxK) per-lane K pair base for pair p (0..7), lane-half hi
__device__ __forceinline__ int kpairA(int p, int hi) {
  return (p < 4) ? (hi * 8 + 2 * p) : (16 + hi * 8 + 2 * (p - 4));
}
// B-operand (32x16 bf16, KxN): same K order as A (table order)
__device__ __forceinline__ int kpairB(int p, int hi) {
  return (p < 4) ? (hi * 8 + 2 * p) : (16 + hi * 8 + 2 * (p - 4));
}

__device__ __forceinline__ void load_opA(Op16& a, const unsigned short* row, int kb, int hi) {
  #pragma unroll
  for (int p = 0; p < 8; ++p)
    a.u[p] = *(const unsigned int*)(row + kb + kpairA(p, hi));
}
__device__ __forceinline__ void load_opB(Op16& a, const unsigned short* row, int kb, int hi) {
  #pragma unroll
  for (int p = 0; p < 8; ++p)
    a.u[p] = *(const unsigned int*)(row + kb + kpairB(p, hi));
}

// Async copy of one 1KB msa row into LDS: each lane moves 16B twice (512B/wave/instr).
// INST_OFFSET is added to BOTH the global and LDS address (ISA 08_async_tensor §4.4).
__device__ __forceinline__ void async_row_to_lds(const float* gsrc, unsigned ldsoff) {
  unsigned long long ga = (unsigned long long)(uintptr_t)gsrc;
  asm volatile(
      "global_load_async_to_lds_b128 %0, %1, off\n\t"
      "global_load_async_to_lds_b128 %0, %1, off offset:512"
      :: "v"(ldsoff), "v"(ga) : "memory");
}

// ---------------------------------------------------------------------------
// Stage 1: per (b,l): async-staged msa stream -> LayerNorm -> hidden
// projection (WMMA, B-operand register-resident) -> outer-product mean (WMMA).
// ---------------------------------------------------------------------------
__global__ __launch_bounds__(128)
void opm_stage1_kernel(const float* __restrict__ msa,
                       const float* __restrict__ mask,
                       const float* __restrict__ gamma,
                       const float* __restrict__ beta,
                       const float* __restrict__ w_hidden,
                       const float* __restrict__ w_out,
                       const float* __restrict__ b_out,
                       float* __restrict__ out,
                       unsigned short* __restrict__ opb,
                       int write_op) {
  __shared__ float raw[2][16 * C_DIM];            // 32KB: msa staging; reused at start as whT image
  __shared__ unsigned short xs[16 * C_DIM];       // 8KB LayerNormed chunk bf16
  __shared__ unsigned short hT[64 * S_N];         // 16KB h^T bf16: hT[h*128+s]
  __shared__ float op_lds[H_DIM * H_DIM];         // 4KB
  __shared__ float g_lds[C_DIM], bt_lds[C_DIM];   // 2KB

  const int l = blockIdx.x;
  const int b = blockIdx.y;
  const int t = threadIdx.x;
  const int lane   = t & 31;
  const int wave   = t >> 5;
  const int laneHi = lane >> 4;
  const int lane15 = lane & 15;

  for (int c = t; c < C_DIM; c += 128) { g_lds[c] = gamma[c]; bt_lds[c] = beta[c]; }

  // ---- Stage w_hidden^T (bf16, u32-packed) into raw[] with coalesced loads ----
  unsigned int* wh32 = (unsigned int*)&raw[0][0];       // [64 n][128 k-pairs]
  for (int i = t; i < 64 * 128; i += 128) {
    const int n  = i & 63;                              // consecutive threads -> consecutive n
    const int k2 = i >> 6;
    wh32[n * 128 + k2] = pack2bf(w_hidden[(size_t)(2 * k2) * 64 + n],
                                 w_hidden[(size_t)(2 * k2 + 1) * 64 + n]);
  }
  __syncthreads();

  // ---- Gather register-resident GEMM1 B-operand (contiguous b128 DS loads) ----
  Op16 bwh[8];
  {
    const int ncol = wave * 16 + lane15;
    #pragma unroll
    for (int kk = 0; kk < 8; ++kk) {
      #pragma unroll
      for (int p = 0; p < 8; ++p)   // (kk*32 + kpairB(p,hi))/2 = kk*16 + (p>>2)*8 + hi*4 + (p&3)
        bwh[kk].u[p] = wh32[ncol * 128 + kk * 16 + (p >> 2) * 8 + laneHi * 4 + (p & 3)];
    }
  }
  __syncthreads();                                      // raw[] free for async staging

  const unsigned raw_base0 = (unsigned)(uintptr_t)&raw[0][0];
  const unsigned raw_base1 = (unsigned)(uintptr_t)&raw[1][0];

  // Prologue: async-stage chunk 0 (each wave fetches its 4 rows).
  #pragma unroll
  for (int r = 0; r < 4; ++r) {
    const int row = wave * 4 + r;
    const float* g = msa + (((size_t)b * S_N + row) * L_N + l) * C_DIM + lane * 4;
    async_row_to_lds(g, raw_base0 + row * 1024 + lane * 16);
  }

  const float inv_c = 1.0f / (float)C_DIM;

  for (int sc = 0; sc < 8; ++sc) {
    const int buf = sc & 1;
    const float* curraw = buf ? raw[1] : raw[0];

    if (sc < 7) {
      const unsigned nxt_base = buf ? raw_base0 : raw_base1;
      #pragma unroll
      for (int r = 0; r < 4; ++r) {
        const int row = wave * 4 + r;
        const int s   = (sc + 1) * 16 + row;
        const float* g = msa + (((size_t)b * S_N + s) * L_N + l) * C_DIM + lane * 4;
        async_row_to_lds(g, nxt_base + row * 1024 + lane * 16);
      }
      asm volatile("s_wait_asynccnt 0x8" ::: "memory");   // current chunk resident
    } else {
      asm volatile("s_wait_asynccnt 0x0" ::: "memory");
    }
    __syncthreads();

    // LayerNorm: each wave handles 4 rows of the 16-row chunk (from LDS staging)
    unsigned int* xs32 = (unsigned int*)xs;
    for (int r = 0; r < 4; ++r) {
      const int row = wave * 4 + r;
      const float* src = curraw + row * C_DIM + lane * 8;
      float v[8];
      float4 p0 = *(const float4*)(src);
      float4 p1 = *(const float4*)(src + 4);
      v[0]=p0.x; v[1]=p0.y; v[2]=p0.z; v[3]=p0.w;
      v[4]=p1.x; v[5]=p1.y; v[6]=p1.z; v[7]=p1.w;
      float sum = 0.f, sq = 0.f;
      #pragma unroll
      for (int j = 0; j < 8; ++j) { sum += v[j]; sq += v[j] * v[j]; }
      #pragma unroll
      for (int m = 16; m >= 1; m >>= 1) {
        sum += __shfl_xor(sum, m, 32);
        sq  += __shfl_xor(sq,  m, 32);
      }
      const float mu   = sum * inv_c;
      const float var  = sq * inv_c - mu * mu;
      const float rstd = rsqrtf(var + 1e-5f);
      float xn[8];
      #pragma unroll
      for (int j = 0; j < 8; ++j) {
        const int c = lane * 8 + j;
        xn[j] = (v[j] - mu) * rstd * g_lds[c] + bt_lds[c];
      }
      #pragma unroll
      for (int q = 0; q < 4; ++q)
        xs32[row * 128 + lane * 4 + q] = pack2bf(xn[2 * q], xn[2 * q + 1]);
    }
    __syncthreads();

    // GEMM1 tile: wave computes N-tile = wave (16 h-cols) x 16 s-rows, K=256
    {
      const unsigned short* arow = xs + lane15 * C_DIM;
      Op16 a0, a1;
      load_opA(a0, arow, 0, laneHi);
      v8f acc = {0.f, 0.f, 0.f, 0.f, 0.f, 0.f, 0.f, 0.f};
      #pragma unroll
      for (int kk = 0; kk < 8; ++kk) {
        if (kk < 7) load_opA(a1, arow, (kk + 1) * 32, laneHi);
        acc = __builtin_amdgcn_wmma_f32_16x16x32_bf16(false, a0.v, false, bwh[kk].v,
                                                      (short)0, acc, false, false);
        if (kk < 7) a0 = a1;
      }
      const int hcol = wave * 16 + lane15;
      unsigned int* hT32 = (unsigned int*)hT;
      const int hbase = hcol * 64 + sc * 8 + laneHi * 4;   // (hcol*128 + sc*16 + laneHi*8)/2
      #pragma unroll
      for (int q = 0; q < 4; ++q)
        hT32[hbase + q] = pack2bf(acc[2 * q], acc[2 * q + 1]);
    }
    __syncthreads();
  }

  // GEMM2: op(32x32) = a^T(32x128) @ bb(128x32) / 128, times mask
  {
    const int dt = wave >> 1, et = wave & 1;
    const unsigned short* arow = hT + (dt * 16 + lane15) * S_N;
    const unsigned short* brow = hT + (32 + et * 16 + lane15) * S_N;
    Op16 a0, b0, a1, b1;
    load_opA(a0, arow, 0, laneHi);
    load_opB(b0, brow, 0, laneHi);
    v8f acc = {0.f, 0.f, 0.f, 0.f, 0.f, 0.f, 0.f, 0.f};
    #pragma unroll
    for (int kk = 0; kk < 4; ++kk) {
      if (kk < 3) {
        load_opA(a1, arow, (kk + 1) * 32, laneHi);
        load_opB(b1, brow, (kk + 1) * 32, laneHi);
      }
      acc = __builtin_amdgcn_wmma_f32_16x16x32_bf16(false, a0.v, false, b0.v,
                                                    (short)0, acc, false, false);
      if (kk < 3) { a0 = a1; b0 = b1; }
    }
    const float scale = mask[(size_t)b * L_N + l] * (1.0f / (float)S_N);
    #pragma unroll
    for (int j = 0; j < 8; ++j) {
      const int d = dt * 16 + j + laneHi * 8;
      const int e = et * 16 + lane15;
      op_lds[d * H_DIM + e] = acc[j] * scale;
    }
  }
  __syncthreads();

  if (write_op) {
    unsigned int* dst32 = (unsigned int*)(opb + ((size_t)b * L_N + l) * K_OP);
    for (int i = t; i < K_OP / 2; i += 128)
      dst32[i] = pack2bf(op_lds[2 * i], op_lds[2 * i + 1]);
  } else {
    #pragma unroll
    for (int r = 0; r < 3; ++r) {
      const int d = t + r * 128;
      float acc = b_out[d];
      for (int k = 0; k < K_OP; ++k)
        acc = fmaf(op_lds[k], w_out[(size_t)k * D_SEQ + d], acc);
      out[((size_t)b * L_N + l) * D_SEQ + d] = acc;
    }
  }
}

// ---------------------------------------------------------------------------
// Stage 2: out(2048x384) = op(2048x1024) @ w_out(1024x384) + b_out, WMMA.
// ---------------------------------------------------------------------------
__global__ __launch_bounds__(256)
void opm_stage2_kernel(const unsigned short* __restrict__ opb,
                       const float* __restrict__ w_out,
                       const float* __restrict__ b_out,
                       float* __restrict__ out) {
  __shared__ unsigned short wT[64 * 128];   // w_out chunk, transposed: wT[nn*128+kk]
  __shared__ unsigned short aT[32 * 128];   // op chunk: aT[mm*128+kk]

  const int t = threadIdx.x;
  const int lane   = t & 31;
  const int wave   = t >> 5;
  const int laneHi = lane >> 4;
  const int lane15 = lane & 15;
  const int m0 = blockIdx.x * 32;
  const int n0 = blockIdx.y * 64;
  const int mt = wave >> 2;
  const int nt = wave & 3;

  unsigned int* wT32 = (unsigned int*)wT;
  unsigned int* aT32 = (unsigned int*)aT;
  const unsigned int* opb32 = (const unsigned int*)opb;

  v8f acc = {0.f, 0.f, 0.f, 0.f, 0.f, 0.f, 0.f, 0.f};

  for (int kc = 0; kc < 8; ++kc) {
    const int k0 = kc * 128;
    // stage w_out chunk transposed, u32-packed along k (coalesced on n)
    for (int i = t; i < 64 * 64; i += 256) {
      const int nn  = i & 63;
      const int kkh = i >> 6;               // k-pair index 0..63
      const int kk  = 2 * kkh;
      wT32[nn * 64 + kkh] = pack2bf(w_out[(size_t)(k0 + kk) * D_SEQ + n0 + nn],
                                    w_out[(size_t)(k0 + kk + 1) * D_SEQ + n0 + nn]);
    }
    // stage op chunk (u32 copies, coalesced)
    for (int i = t; i < 32 * 64; i += 256) {
      const int kkh = i & 63;
      const int mm  = i >> 6;
      aT32[mm * 64 + kkh] = opb32[(size_t)(m0 + mm) * (K_OP / 2) + kc * 64 + kkh];
    }
    __syncthreads();

    const unsigned short* arow = aT + (mt * 16 + lane15) * 128;
    const unsigned short* brow = wT + (nt * 16 + lane15) * 128;
    Op16 a0, b0, a1, b1;
    load_opA(a0, arow, 0, laneHi);
    load_opB(b0, brow, 0, laneHi);
    #pragma unroll
    for (int ks = 0; ks < 4; ++ks) {
      if (ks < 3) {
        load_opA(a1, arow, (ks + 1) * 32, laneHi);
        load_opB(b1, brow, (ks + 1) * 32, laneHi);
      }
      acc = __builtin_amdgcn_wmma_f32_16x16x32_bf16(false, a0.v, false, b0.v,
                                                    (short)0, acc, false, false);
      if (ks < 3) { a0 = a1; b0 = b1; }
    }
    __syncthreads();
  }

  #pragma unroll
  for (int j = 0; j < 8; ++j) {
    const int row = m0 + mt * 16 + j + laneHi * 8;
    const int col = n0 + nt * 16 + lane15;
    out[(size_t)row * D_SEQ + col] = acc[j] + b_out[col];
  }
}

extern "C" void kernel_launch(void* const* d_in, const int* in_sizes, int n_in,
                              void* d_out, int out_size, void* d_ws, size_t ws_size,
                              hipStream_t stream) {
  (void)in_sizes; (void)n_in; (void)out_size;
  const float* msa      = (const float*)d_in[0];
  const float* mask     = (const float*)d_in[1];
  const float* gamma    = (const float*)d_in[2];
  const float* beta     = (const float*)d_in[3];
  const float* w_hidden = (const float*)d_in[4];
  const float* w_out    = (const float*)d_in[5];
  const float* b_out    = (const float*)d_in[6];
  float* out = (float*)d_out;

  const size_t op_bytes = (size_t)B_N * L_N * K_OP * sizeof(unsigned short); // 4 MB
  const int use_ws = (d_ws != nullptr && ws_size >= op_bytes) ? 1 : 0;
  unsigned short* opb = (unsigned short*)d_ws;

  dim3 grid1(L_N, B_N);
  dim3 block1(128);
  opm_stage1_kernel<<<grid1, block1, 0, stream>>>(msa, mask, gamma, beta,
                                                  w_hidden, w_out, b_out, out,
                                                  opb, use_ws);
  if (use_ws) {
    dim3 grid2((B_N * L_N) / 32, D_SEQ / 64);
    dim3 block2(256);
    opm_stage2_kernel<<<grid2, block2, 0, stream>>>(opb, w_out, b_out, out);
  }
}
